// Attention_81458349736716
// MI455X (gfx1250) — matrix operand order repaired and checked by reading, not compiled, not run on hardware
//
#include <hip/hip_runtime.h>
#include <math.h>

typedef __attribute__((ext_vector_type(16))) _Float16 v16h;
typedef __attribute__((ext_vector_type(8)))  _Float16 v8h;
typedef __attribute__((ext_vector_type(8)))  float    v8f;
typedef __attribute__((ext_vector_type(4)))  unsigned int u32x4;
typedef __attribute__((ext_vector_type(8)))  unsigned int u32x8;

#define DIM   1024
#define HEADS 16
#define DHEAD 64
#define BATCH 2
#define SEQ   2048
#define NKPAD 2112        // 2049 kv rows padded to 33*64
#define NPOS  4095        // 2*SEQ-1

// ---------------------------------------------------------------------------
// Tensor Data Mover: 2D tile (global, row-major, f16) -> LDS, with LDS row
// padding. Descriptor per CDNA5 ISA (D# group0/group1):
//   g0: [1:0]=count=1, [63:32]=lds_addr, [120:64]=global_addr, [127:126]=type=2
//   g1: [17:16]=data_size(1=2B), [20]=pad_en, [24:22]=pad_interval,
//       [31:25]=pad_amount, [79:48]=tensor_dim0, [111:80]=tensor_dim1,
//       [127:112]=tile_dim0, [143:128]=tile_dim1, [207:160]=dim0_stride
// All operands are wave-uniform; issued by one wave, completion via TENSORcnt.
// ---------------------------------------------------------------------------
__device__ __forceinline__ void tdm_load_2d(
    unsigned lds_addr, const void* gaddr,
    unsigned tensor_d0, unsigned tensor_d1,
    unsigned tile_d0, unsigned tile_d1,
    unsigned stride0, unsigned pad_interval, unsigned pad_amount)
{
  unsigned long long ga = (unsigned long long)(uintptr_t)gaddr;
  u32x4 g0;
  g0[0] = 1u;                                                  // count=1
  g0[1] = lds_addr;                                            // LDS byte addr
  g0[2] = (unsigned)ga;                                        // gaddr[31:0]
  g0[3] = (unsigned)((ga >> 32) & 0x01ffffffu) | (2u << 30);   // [56:32]|type=2
  u32x8 g1;
  g1[0] = (1u << 16) | (1u << 20)                              // 2B, pad_en
        | ((pad_interval & 7u) << 22) | ((pad_amount & 127u) << 25);
  g1[1] = (tensor_d0 & 0xffffu) << 16;
  g1[2] = ((tensor_d0 >> 16) & 0xffffu) | ((tensor_d1 & 0xffffu) << 16);
  g1[3] = ((tensor_d1 >> 16) & 0xffffu) | ((tile_d0 & 0xffffu) << 16);
  g1[4] = (tile_d1 & 0xffffu);                                 // tile_dim2=0
  g1[5] = stride0;                                             // stride[31:0]
  g1[6] = 0u;
  g1[7] = 0u;
  asm volatile("tensor_load_to_lds %0, %1" :: "s"(g0), "s"(g1) : "memory");
}

// ---------------------------------------------------------------------------
// fp32 -> fp16 convert
// ---------------------------------------------------------------------------
__global__ void cvt_f16(const float* __restrict__ s, _Float16* __restrict__ d, int n) {
  int i = blockIdx.x * 256 + threadIdx.x;
  if (i < n) d[i] = (_Float16)s[i];
}

// ---------------------------------------------------------------------------
// Row LayerNorm over 1024 cols. posmode: row value = (r-2047)*w0[c]+b0[c].
// Optional SiLU. Writes f16 (outh) or f32 (outf).
// ---------------------------------------------------------------------------
__global__ __launch_bounds__(256) void ln_rows(
    const float* __restrict__ X, const float* __restrict__ w0,
    const float* __restrict__ b0, const float* __restrict__ g,
    const float* __restrict__ bta, int R, int posmode, int do_silu,
    _Float16* __restrict__ outh, float* __restrict__ outf)
{
  __shared__ float red[2][8];
  int tid = threadIdx.x;
  int r = blockIdx.x;
  if (r >= R) return;
  float v[4];
  float s = 0.f, q = 0.f;
#pragma unroll
  for (int u = 0; u < 4; ++u) {
    int c = tid + u * 256;
    float val;
    if (posmode) { float p = (float)r - 2047.0f; val = p * w0[c] + b0[c]; }
    else          val = X[(size_t)r * DIM + c];
    v[u] = val; s += val; q += val * val;
  }
#pragma unroll
  for (int o = 16; o > 0; o >>= 1) { s += __shfl_down(s, o); q += __shfl_down(q, o); }
  int wave = tid >> 5, lane = tid & 31;
  if (lane == 0) { red[0][wave] = s; red[1][wave] = q; }
  __syncthreads();
  if (tid == 0) {
    float ss = 0.f, qq = 0.f;
#pragma unroll
    for (int w = 0; w < 8; ++w) { ss += red[0][w]; qq += red[1][w]; }
    red[0][0] = ss; red[1][0] = qq;
  }
  __syncthreads();
  float mean = red[0][0] * (1.0f / 1024.0f);
  float var  = red[1][0] * (1.0f / 1024.0f) - mean * mean;
  float rs   = rsqrtf(var + 1e-5f);
#pragma unroll
  for (int u = 0; u < 4; ++u) {
    int c = tid + u * 256;
    float y = (v[u] - mean) * rs * g[c] + bta[c];
    if (do_silu) y = y / (1.0f + __expf(-y));
    if (outh) outh[(size_t)r * DIM + c] = (_Float16)y;
    else      outf[(size_t)r * DIM + c] = y;
  }
}

// ---------------------------------------------------------------------------
// Generic f16 WMMA GEMM: C(M,N) = A(M,K) * B(K,N), row-major f16 inputs.
// Block 256 thr = 8 waves; tile 128x64, K-step 32; each wave -> 16x64 output.
// A tile staged via TDM (zero-fill OOB rows), B tile via vector loads + LDS
// transpose to N-major.
// mode 0: write q f16, *0.125, layout (b,h,n,64)
// mode 1: write k/v f16 to padded (b,2112,64) at row 1+i
// mode 2: write f32 (+optional bias[col]) to outf row-major N
// ---------------------------------------------------------------------------
#define GBM 128
#define GBN 64
#define GBK 32
#define GLDA 40
#define GLDB 40

__global__ __launch_bounds__(256) void gemm_f16(
    const _Float16* __restrict__ A, const _Float16* __restrict__ Bw,
    int M, int Nn, int K, const float* __restrict__ bias, int mode,
    _Float16* __restrict__ outh, _Float16* __restrict__ outh2,
    float* __restrict__ outf)
{
  __shared__ _Float16 As[GBM * GLDA];
  __shared__ _Float16 Bs[GBN * GLDB];   // N-major: Bs[n*GLDB + k]
  int tid = threadIdx.x;
  int wave = tid >> 5, lane = tid & 31;
  int rowBase = blockIdx.y * GBM, colBase = blockIdx.x * GBN;

  v8f zz = {};
  v8f acc[4]; acc[0] = zz; acc[1] = zz; acc[2] = zz; acc[3] = zz;

  int lb_k = tid >> 3, lb_n = (tid & 7) * 8;
  int n15 = lane & 15;
  int akb = (lane < 16) ? 0 : 8;   // A frag K-base per half-wave
  int bkb = (lane < 16) ? 0 : 8;   // B frag K-base per half-wave

  for (int k0 = 0; k0 < K; k0 += GBK) {
    // A tile (128x32) via Tensor Data Mover; LDS rows padded 32+8 halves.
    // pad_interval=3 -> every 16 DWORDs (32 halves), pad_amount=3 -> 4 DWORDs.
    if (wave == 0) {
      tdm_load_2d((unsigned)(uintptr_t)&As[0],
                  A + (size_t)rowBase * K + k0,
                  (unsigned)(K - k0), (unsigned)(M - rowBase),
                  GBK, GBM, (unsigned)K, 3u, 3u);
    }
    { // load B tile (32x64), transpose to N-major
      const v8h* src = (const v8h*)(Bw + (size_t)(k0 + lb_k) * Nn + colBase + lb_n);
      v8h bv = src[0];
#pragma unroll
      for (int j = 0; j < 8; ++j) Bs[(lb_n + j) * GLDB + lb_k] = bv[j];
      if (k0 + GBK < K)   // prefetch next B tile through L2
        __builtin_prefetch(Bw + (size_t)(k0 + GBK + lb_k) * Nn + colBase + lb_n, 0, 1);
    }
    if (wave == 0) __builtin_amdgcn_s_wait_tensorcnt(0);
    __syncthreads();
    v16h afr;
    {
      const _Float16* ab = &As[(wave * 16 + n15) * GLDA + akb];
#pragma unroll
      for (int e = 0; e < 8; ++e) { afr[e] = ab[e]; afr[8 + e] = ab[16 + e]; }
    }
#pragma unroll
    for (int t = 0; t < 4; ++t) {
      const _Float16* bb = &Bs[(t * 16 + n15) * GLDB + bkb];
      v16h bfr;
#pragma unroll
      for (int e = 0; e < 8; ++e) { bfr[e] = bb[e]; bfr[8 + e] = bb[16 + e]; }
      acc[t] = __builtin_amdgcn_wmma_f32_16x16x32_f16(
          false, afr, false, bfr, (short)0, acc[t], false, false);
    }
    __syncthreads();
  }

  int mh = (lane & 16) ? 8 : 0;
#pragma unroll
  for (int t = 0; t < 4; ++t) {
#pragma unroll
    for (int e = 0; e < 8; ++e) {
      int gr = rowBase + wave * 16 + e + mh;
      int gc = colBase + t * 16 + n15;
      if (gr >= M) continue;
      float val = acc[t][e];
      if (mode == 0) {
        int b = gr >> 11, i = gr & 2047, hh = gc >> 6, d = gc & 63;
        outh[((size_t)((b * HEADS + hh) * SEQ + i)) * DHEAD + d] =
            (_Float16)(val * 0.125f);
      } else if (mode == 1) {
        int b = gr >> 11, i = gr & 2047;
        if (gc < DHEAD)
          outh [((size_t)(b * NKPAD + 1 + i)) * DHEAD + gc] = (_Float16)val;
        else
          outh2[((size_t)(b * NKPAD + 1 + i)) * DHEAD + (gc - DHEAD)] = (_Float16)val;
      } else {
        float vv = val + (bias ? bias[gc] : 0.0f);
        outf[(size_t)gr * Nn + gc] = vv;
      }
    }
  }
}

// ---------------------------------------------------------------------------
// null k/v prepend + zero pad rows 2049..2111
// ---------------------------------------------------------------------------
__global__ void fill_null(const float* __restrict__ null_kv,
                          _Float16* __restrict__ kh, _Float16* __restrict__ vh)
{
  int t = threadIdx.x;
  for (int b = 0; b < BATCH; ++b) {
    size_t base = (size_t)b * NKPAD * DHEAD;
    if (t < DHEAD) {
      kh[base + t] = (_Float16)null_kv[t];
      vh[base + t] = (_Float16)null_kv[DHEAD + t];
    }
    for (int idx = t; idx < (NKPAD - 2049) * DHEAD; idx += 256) {
      kh[base + 2049 * DHEAD + idx] = (_Float16)0.0f;
      vh[base + 2049 * DHEAD + idx] = (_Float16)0.0f;
    }
  }
}

// ---------------------------------------------------------------------------
// pos-bias final projection: pbh[p,h] = h1[p,:] . w2[:,h] + b2[h]  (4095x16)
// ---------------------------------------------------------------------------
__global__ __launch_bounds__(256) void pb2_kernel(
    const _Float16* __restrict__ h1, const float* __restrict__ w2,
    const float* __restrict__ b2, float* __restrict__ pbh)
{
  int t = threadIdx.x;
  int row = blockIdx.x * 16 + (t >> 4);
  int col = t & 15;
  if (row >= NPOS) return;
  float acc = b2[col];
  for (int k = 0; k < DIM; ++k)
    acc += (float)h1[(size_t)row * DIM + k] * w2[k * HEADS + col];
  pbh[row * HEADS + col] = acc;
}

// ---------------------------------------------------------------------------
// Flash attention. Block = 8 waves, 128 query rows, (b,h) per blockIdx.y.
// K tile staged by TDM; V tile loaded+transposed by the block. 64-key tiles;
// S via 8 WMMAs, online softmax, P@V via 8 WMMAs. Scores biased with
// null_attn_bias (j==0) / pbh gather (j>0), causal j<=i+1.
// ---------------------------------------------------------------------------
__global__ __launch_bounds__(256) void attn_kernel(
    const _Float16* __restrict__ qh, const _Float16* __restrict__ kh,
    const _Float16* __restrict__ vh, const float* __restrict__ pbh,
    const float* __restrict__ nullbias, _Float16* __restrict__ oh)
{
  __shared__ _Float16 Qs[128 * 72];      // [qrow][d]
  __shared__ _Float16 Ks[64 * 72];       // [key][d]   (B n-major for Q@K^T)
  __shared__ _Float16 Vs[64 * 72];       // [d][key]   (B n-major for P@V)
  __shared__ _Float16 Ps[8 * 16 * 72];   // per-wave P tile [16][64]

  int tid = threadIdx.x, wave = tid >> 5, lane = tid & 31;
  int bh = blockIdx.y, b = bh >> 4, h = bh & 15;
  int q0 = blockIdx.x * 128;
  const _Float16* Q  = qh + ((size_t)bh * SEQ + q0) * DHEAD;
  const _Float16* Kp = kh + (size_t)b * NKPAD * DHEAD;
  const _Float16* Vp = vh + (size_t)b * NKPAD * DHEAD;

  { // load Q tile once
    int row = tid >> 1, seg = (tid & 1) * 32;
    const v8h* src = (const v8h*)(Q + (size_t)row * DHEAD + seg);
#pragma unroll
    for (int j = 0; j < 4; ++j) *(v8h*)(&Qs[row * 72 + seg + j * 8]) = src[j];
  }
  __syncthreads();

  int n15 = lane & 15;
  int mh  = (lane & 16) ? 8 : 0;
  int akb = (lane < 16) ? 0 : 8;
  int bkb = (lane < 16) ? 0 : 8;

  v16h qa[2];
#pragma unroll
  for (int c = 0; c < 2; ++c) {
    const _Float16* ab = &Qs[(wave * 16 + n15) * 72 + c * 32 + akb];
#pragma unroll
    for (int e = 0; e < 8; ++e) { qa[c][e] = ab[e]; qa[c][8 + e] = ab[16 + e]; }
  }

  v8f zz = {};
  v8f Oacc[4]; Oacc[0] = zz; Oacc[1] = zz; Oacc[2] = zz; Oacc[3] = zz;
  float m_run[8], l_run[8];
#pragma unroll
  for (int e = 0; e < 8; ++e) { m_run[e] = -3.0e38f; l_run[e] = 0.0f; }
  float nb = nullbias[h];

  int ntiles = q0 / 64 + 3;         // ceil((q0+129)/64)
  if (ntiles > 33) ntiles = 33;

  for (int kt = 0; kt < ntiles; ++kt) {
    int jb = kt * 64;
    // K tile (64x64) via TDM: pad_interval=4 (32 DWORDs/row), pad 4 DWORDs.
    if (wave == 0) {
      tdm_load_2d((unsigned)(uintptr_t)&Ks[0],
                  Kp + (size_t)jb * DHEAD,
                  DHEAD, (unsigned)(NKPAD - jb),
                  DHEAD, 64, DHEAD, 4u, 3u);
    }
    { // V tile loaded + transposed by all threads
      int row = tid >> 2, seg = (tid & 3) * 16;
      const v8h* vsrc = (const v8h*)(Vp + (size_t)(jb + row) * DHEAD + seg);
      v8h v0 = vsrc[0], v1 = vsrc[1];
#pragma unroll
      for (int j = 0; j < 8; ++j) {
        Vs[(seg + j)     * 72 + row] = v0[j];
        Vs[(seg + 8 + j) * 72 + row] = v1[j];
      }
      if (kt + 1 < ntiles) { // prefetch next K/V tiles
        __builtin_prefetch(Kp + (size_t)(jb + 64 + row) * DHEAD + seg, 0, 1);
        __builtin_prefetch(Vp + (size_t)(jb + 64 + row) * DHEAD + seg, 0, 1);
      }
    }
    if (wave == 0) __builtin_amdgcn_s_wait_tensorcnt(0);
    __syncthreads();

    // S = Q @ K^T  (16 q-rows x 64 keys per wave)
    v8f s[4]; s[0] = zz; s[1] = zz; s[2] = zz; s[3] = zz;
#pragma unroll
    for (int t = 0; t < 4; ++t) {
#pragma unroll
      for (int c = 0; c < 2; ++c) {
        const _Float16* bb = &Ks[(t * 16 + n15) * 72 + c * 32 + bkb];
        v16h bfr;
#pragma unroll
        for (int e = 0; e < 8; ++e) { bfr[e] = bb[e]; bfr[8 + e] = bb[16 + e]; }
        s[t] = __builtin_amdgcn_wmma_f32_16x16x32_f16(
            false, qa[c], false, bfr, (short)0, s[t], false, false);
      }
    }

    // bias + causal mask
#pragma unroll
    for (int t = 0; t < 4; ++t) {
#pragma unroll
      for (int e = 0; e < 8; ++e) {
        int i = q0 + wave * 16 + e + mh;
        int j = jb + t * 16 + n15;
        float sv = s[t][e];
        if (j > i + 1 || j >= 2049) sv = -1.0e30f;
        else sv += (j == 0) ? nb : pbh[(i - j + 2048) * HEADS + h];
        s[t][e] = sv;
      }
    }

    // online softmax
    float mnew[8], rsum[8];
#pragma unroll
    for (int e = 0; e < 8; ++e) {
      float tm = fmaxf(fmaxf(s[0][e], s[1][e]), fmaxf(s[2][e], s[3][e]));
#pragma unroll
      for (int o = 1; o < 16; o <<= 1) tm = fmaxf(tm, __shfl_xor(tm, o));
      mnew[e] = fmaxf(m_run[e], tm);
      rsum[e] = 0.0f;
    }
    _Float16* Pw = &Ps[wave * 16 * 72];
#pragma unroll
    for (int t = 0; t < 4; ++t) {
#pragma unroll
      for (int e = 0; e < 8; ++e) {
        float p = __expf(s[t][e] - mnew[e]);
        rsum[e] += p;
        Pw[(e + mh) * 72 + t * 16 + n15] = (_Float16)p;
      }
    }
#pragma unroll
    for (int e = 0; e < 8; ++e) {
      float rs = rsum[e];
#pragma unroll
      for (int o = 1; o < 16; o <<= 1) rs += __shfl_xor(rs, o);
      float alpha = __expf(m_run[e] - mnew[e]);
      l_run[e] = l_run[e] * alpha + rs;
      m_run[e] = mnew[e];
#pragma unroll
      for (int dt = 0; dt < 4; ++dt) Oacc[dt][e] *= alpha;
    }

    // O += P @ V
    v16h pa[2];
#pragma unroll
    for (int c = 0; c < 2; ++c) {
      const _Float16* ab = &Pw[n15 * 72 + c * 32 + akb];
#pragma unroll
      for (int e = 0; e < 8; ++e) { pa[c][e] = ab[e]; pa[c][8 + e] = ab[16 + e]; }
    }
#pragma unroll
    for (int dt = 0; dt < 4; ++dt) {
#pragma unroll
      for (int c = 0; c < 2; ++c) {
        const _Float16* bb = &Vs[(dt * 16 + n15) * 72 + c * 32 + bkb];
        v16h bfr;
#pragma unroll
        for (int e = 0; e < 8; ++e) { bfr[e] = bb[e]; bfr[8 + e] = bb[16 + e]; }
        Oacc[dt] = __builtin_amdgcn_wmma_f32_16x16x32_f16(
            false, pa[c], false, bfr, (short)0, Oacc[dt], false, false);
      }
    }
    __syncthreads();
  }

  // write O / l  to (b, n, h*64+d) f16 for the Wo GEMM
#pragma unroll
  for (int e = 0; e < 8; ++e) {
    float inv = 1.0f / l_run[e];
    int i = q0 + wave * 16 + e + mh;
#pragma unroll
    for (int dt = 0; dt < 4; ++dt) {
      int d = dt * 16 + n15;
      oh[((size_t)(b * SEQ + i)) * DIM + h * DHEAD + d] =
          (_Float16)(Oacc[dt][e] * inv);
    }
  }
}

// ---------------------------------------------------------------------------
// host launcher
// ---------------------------------------------------------------------------
extern "C" void kernel_launch(void* const* d_in, const int* in_sizes, int n_in,
                              void* d_out, int out_size, void* d_ws, size_t ws_size,
                              hipStream_t stream) {
  const float* x        = (const float*)d_in[0];
  const float* ln_g     = (const float*)d_in[1];
  const float* ln_b     = (const float*)d_in[2];
  const float* Wq       = (const float*)d_in[3];
  const float* Wkv      = (const float*)d_in[4];
  const float* null_kv  = (const float*)d_in[5];
  const float* null_ab  = (const float*)d_in[6];
  const float* pb_w0    = (const float*)d_in[7];
  const float* pb_b0    = (const float*)d_in[8];
  const float* pb_ln0_g = (const float*)d_in[9];
  const float* pb_ln0_b = (const float*)d_in[10];
  const float* pb_w1    = (const float*)d_in[11];
  const float* pb_b1    = (const float*)d_in[12];
  const float* pb_ln1_g = (const float*)d_in[13];
  const float* pb_ln1_b = (const float*)d_in[14];
  const float* pb_w2    = (const float*)d_in[15];
  const float* pb_b2    = (const float*)d_in[16];
  const float* Wo       = (const float*)d_in[17];
  const float* o_ln_g   = (const float*)d_in[18];
  const float* o_ln_b   = (const float*)d_in[19];
  (void)in_sizes; (void)n_in; (void)out_size; (void)ws_size;

  size_t off = 0;
  auto alloc = [&](size_t bytes) -> void* {
    void* p = (char*)d_ws + off;
    off += (bytes + 255) & ~(size_t)255;
    return p;
  };
  const int ROWS = BATCH * SEQ;  // 4096
  _Float16* xnh  = (_Float16*)alloc((size_t)ROWS * DIM * 2);
  _Float16* qh   = (_Float16*)alloc((size_t)BATCH * HEADS * SEQ * DHEAD * 2);
  _Float16* kh   = (_Float16*)alloc((size_t)BATCH * NKPAD * DHEAD * 2);
  _Float16* vh   = (_Float16*)alloc((size_t)BATCH * NKPAD * DHEAD * 2);
  _Float16* h0h  = (_Float16*)alloc((size_t)NPOS * DIM * 2);
  float*    pbh  = (float*)   alloc((size_t)NPOS * HEADS * 4);
  _Float16* oh   = (_Float16*)alloc((size_t)ROWS * DIM * 2);
  float*    yf   = (float*)   alloc((size_t)ROWS * DIM * 4);
  _Float16* wqh  = (_Float16*)alloc((size_t)DIM * DIM * 2);
  _Float16* wkvh = (_Float16*)alloc((size_t)DIM * 2 * DHEAD * 2);
  _Float16* w1h  = (_Float16*)alloc((size_t)DIM * DIM * 2);
  _Float16* woh  = (_Float16*)alloc((size_t)DIM * DIM * 2);

  // weights -> f16
  cvt_f16<<<(DIM * DIM + 255) / 256, 256, 0, stream>>>(Wq, wqh, DIM * DIM);
  cvt_f16<<<(DIM * 2 * DHEAD + 255) / 256, 256, 0, stream>>>(Wkv, wkvh, DIM * 2 * DHEAD);
  cvt_f16<<<(DIM * DIM + 255) / 256, 256, 0, stream>>>(pb_w1, w1h, DIM * DIM);
  cvt_f16<<<(DIM * DIM + 255) / 256, 256, 0, stream>>>(Wo, woh, DIM * DIM);

  // xn = LN(x) -> f16
  ln_rows<<<ROWS, 256, 0, stream>>>(x, nullptr, nullptr, ln_g, ln_b,
                                    ROWS, 0, 0, xnh, nullptr);
  // q = (xn @ Wq) * scale  -> (b,h,n,64) f16
  gemm_f16<<<dim3(DIM / 64, ROWS / 128), 256, 0, stream>>>(
      xnh, wqh, ROWS, DIM, DIM, nullptr, 0, qh, nullptr, nullptr);
  // kv = xn @ Wkv -> k,v padded f16
  gemm_f16<<<dim3(2, ROWS / 128), 256, 0, stream>>>(
      xnh, wkvh, ROWS, 2 * DHEAD, DIM, nullptr, 1, kh, vh, nullptr);
  fill_null<<<1, 256, 0, stream>>>(null_kv, kh, vh);

  // pos-bias MLP
  ln_rows<<<NPOS, 256, 0, stream>>>(nullptr, pb_w0, pb_b0, pb_ln0_g, pb_ln0_b,
                                    NPOS, 1, 1, h0h, nullptr);
  gemm_f16<<<dim3(DIM / 64, (NPOS + 127) / 128), 256, 0, stream>>>(
      h0h, w1h, NPOS, DIM, DIM, pb_b1, 2, nullptr, nullptr, yf);
  ln_rows<<<NPOS, 256, 0, stream>>>(yf, nullptr, nullptr, pb_ln1_g, pb_ln1_b,
                                    NPOS, 0, 1, h0h, nullptr);
  pb2_kernel<<<(NPOS + 15) / 16, 256, 0, stream>>>(h0h, pb_w2, pb_b2, pbh);

  // flash attention
  attn_kernel<<<dim3(SEQ / 128, BATCH * HEADS), 256, 0, stream>>>(
      qh, kh, vh, pbh, null_ab, oh);

  // out @ Wo -> f32, then final LN -> d_out
  gemm_f16<<<dim3(DIM / 64, ROWS / 128), 256, 0, stream>>>(
      oh, woh, ROWS, DIM, DIM, nullptr, 2, nullptr, nullptr, yf);
  ln_rows<<<ROWS, 256, 0, stream>>>(yf, nullptr, nullptr, o_ln_g, o_ln_b,
                                    ROWS, 0, 0, nullptr, (float*)d_out);
}
